// FocusAttention_64639257804908
// MI455X (gfx1250) — matrix operand order repaired and checked by reading, not compiled, not run on hardware
//
#include <hip/hip_runtime.h>
#include <hip/hip_bf16.h>

// FocusAttention fused kernel for MI455X (gfx1250, wave32).
//
// Shapes: B=8, L=1024, H=8, E=64, S=1024, D=64.
// Grid: one workgroup per (b, h, 64-query tile) = 1024 WGs, 256 threads (8 waves).
// Flash-style two-pass: pass 1 computes row stats (sum x^2, sum x^4, max x^2) of the
// score rows from WMMA accumulators; pass 2 recomputes scores, applies
// exp(c*(x^2 - max)) with c = sqrt(sum x^2)/sqrt(sum x^4), and runs the P*V GEMM.
// Score tensor (256 MB if materialized) never touches HBM. LDS ~64 KB/WG.

typedef __attribute__((ext_vector_type(16))) _Float16 v16h;
typedef __attribute__((ext_vector_type(8)))  float    v8f;

#define B_ 8
#define L_ 1024
#define H_ 8
#define E_ 64
#define S_ 1024
#define D_ 64

#define QTILE   64
#define KCHUNK  128
#define NCHUNKS (S_ / KCHUNK)

// LDS row strides in halfwords; +8 halves (16B) padding breaks the 256B bank period.
#define RSQ 72
#define RSK 72
#define RSV 136
#define RSP 136

#define WMMA_F16(A, Bm, C) \
  __builtin_amdgcn_wmma_f32_16x16x32_f16(false, (A), false, (Bm), (short)0, (C), false, false)

// A-fragment (16x32 f16, ISA 7.12.2): lane l holds row m=l&15; halves 0-7 are
// K = (l>>4)*8 + 0..7, halves 8-15 are K = 16 + (l>>4)*8 + 0..7.
static __device__ __forceinline__ v16h afrag(const _Float16* base, int rs, int row,
                                             int kbase, int lane) {
  const int h2 = lane >> 4;
  const _Float16* p = base + row * rs + kbase + h2 * 8;
  v16h r;
  reinterpret_cast<uint4*>(&r)[0] = reinterpret_cast<const uint4*>(p)[0];
  reinterpret_cast<uint4*>(&r)[1] = reinterpret_cast<const uint4*>(p + 16)[0];
  return r;
}

// B-fragment (32x16 f16): lane l holds column n=l&15; halves 0-7 are
// K = (l>>4)*8 + 0..7, halves 8-15 are K = 16 + (l>>4)*8 + 0..7. `base` stores B^T row-major
// (row n of base == column n of B), so this is two 16B reads.
static __device__ __forceinline__ v16h bfrag(const _Float16* base, int rs, int col,
                                             int kbase, int lane) {
  const int kh = lane >> 4;
  const _Float16* p = base + col * rs + kbase + kh * 8;
  v16h r;
  reinterpret_cast<uint4*>(&r)[0] = reinterpret_cast<const uint4*>(p)[0];
  reinterpret_cast<uint4*>(&r)[1] = reinterpret_cast<const uint4*>(p + 16)[0];
  return r;
}

__global__ __launch_bounds__(256)
void focus_attn_kernel(const float* __restrict__ Q, const float* __restrict__ K,
                       const float* __restrict__ V, float* __restrict__ O) {
  __shared__ __align__(16) _Float16 sQ [QTILE  * RSQ];   //  9.0 KB  phi_p(Q) f16
  __shared__ __align__(16) _Float16 sK [KCHUNK * RSK];   // 18.0 KB  phi_p(K) chunk f16
  __shared__ __align__(16) _Float16 sVt[D_     * RSV];   // 17.0 KB  V chunk, transposed f16
  __shared__ __align__(16) _Float16 sP [QTILE  * RSP];   // 17.0 KB  exp'd scores f16
  __shared__ float sS2[QTILE], sS4[QTILE], sC[QTILE], sZ[QTILE];
  __shared__ int   sMx[QTILE];                            // max(x^2) as int (x^2 >= 0)

  const int tid  = threadIdx.x;
  const int lane = tid & 31;
  const int wave = tid >> 5;
  const int tile = blockIdx.x & (L_ / QTILE - 1);
  const int h    = (blockIdx.x >> 4) & (H_ - 1);
  const int b    = blockIdx.x >> 7;
  const int l0   = tile * QTILE;

  // ---- phi_p(Q tile) -> sQ : 4 threads per row, 16 elems each ----
  {
    const int row = tid >> 2, qq = tid & 3;
    const float* qp = Q + (((size_t)b * L_ + l0 + row) * H_ + h) * E_ + qq * 16;
    float r[16];
    #pragma unroll
    for (int i = 0; i < 4; ++i) {
      float4 v = reinterpret_cast<const float4*>(qp)[i];
      r[i * 4 + 0] = fmaxf(v.x, 0.f); r[i * 4 + 1] = fmaxf(v.y, 0.f);
      r[i * 4 + 2] = fmaxf(v.z, 0.f); r[i * 4 + 3] = fmaxf(v.w, 0.f);
    }
    float s2 = 0.f, s4 = 0.f;
    #pragma unroll
    for (int j = 0; j < 16; ++j) { float x2 = r[j] * r[j]; s2 += x2; s4 += x2 * x2; }
    s2 += __shfl_xor(s2, 1, 32); s4 += __shfl_xor(s4, 1, 32);
    s2 += __shfl_xor(s2, 2, 32); s4 += __shfl_xor(s4, 2, 32);
    const float scale = sqrtf(s2) * rsqrtf(s4);   // norm(x)/norm(x^2)
    union { _Float16 hh[8]; uint4 u; } pk;
    #pragma unroll
    for (int i = 0; i < 2; ++i) {
      #pragma unroll
      for (int j = 0; j < 8; ++j) { float t = r[i * 8 + j]; pk.hh[j] = (_Float16)(t * t * scale); }
      reinterpret_cast<uint4*>(&sQ[row * RSQ + qq * 16 + i * 8])[0] = pk.u;
    }
  }
  if (tid < QTILE) { sS2[tid] = 0.f; sS4[tid] = 0.f; sZ[tid] = 0.f; sMx[tid] = 0; }
  __syncthreads();

  // ---- chunk loaders: 2 threads per row, 32 elems each ----
  auto loadK = [&](int s0) {
    const int row = tid >> 1, hf = tid & 1;
    const float* kp = K + (((size_t)b * S_ + s0 + row) * H_ + h) * E_ + hf * 32;
    float r[32];
    #pragma unroll
    for (int i = 0; i < 8; ++i) {
      float4 v = reinterpret_cast<const float4*>(kp)[i];
      r[i * 4 + 0] = fmaxf(v.x, 0.f); r[i * 4 + 1] = fmaxf(v.y, 0.f);
      r[i * 4 + 2] = fmaxf(v.z, 0.f); r[i * 4 + 3] = fmaxf(v.w, 0.f);
    }
    float s2 = 0.f, s4 = 0.f;
    #pragma unroll
    for (int j = 0; j < 32; ++j) { float x2 = r[j] * r[j]; s2 += x2; s4 += x2 * x2; }
    s2 += __shfl_xor(s2, 1, 32); s4 += __shfl_xor(s4, 1, 32);
    const float scale = sqrtf(s2) * rsqrtf(s4);
    union { _Float16 hh[8]; uint4 u; } pk;
    #pragma unroll
    for (int i = 0; i < 4; ++i) {
      #pragma unroll
      for (int j = 0; j < 8; ++j) { float t = r[i * 8 + j]; pk.hh[j] = (_Float16)(t * t * scale); }
      reinterpret_cast<uint4*>(&sK[row * RSK + hf * 32 + i * 8])[0] = pk.u;
    }
  };
  auto loadV = [&](int s0) {
    const int row = tid >> 1, hf = tid & 1;
    const float* vp = V + (((size_t)b * S_ + s0 + row) * H_ + h) * D_ + hf * 32;
    #pragma unroll
    for (int i = 0; i < 8; ++i) {
      float4 v = reinterpret_cast<const float4*>(vp)[i];
      const int d0 = hf * 32 + i * 4;
      sVt[(d0 + 0) * RSV + row] = (_Float16)v.x;
      sVt[(d0 + 1) * RSV + row] = (_Float16)v.y;
      sVt[(d0 + 2) * RSV + row] = (_Float16)v.z;
      sVt[(d0 + 3) * RSV + row] = (_Float16)v.w;
    }
  };

  // wave -> (mi, nj): rows m0..m0+15, key columns nj*64..+63 (4 N-tiles)
  const int mi = wave & 3, nj = wave >> 2;
  const int m0 = mi << 4;
  const int l15 = lane & 15, lhi = (lane >> 4) << 3;

  auto gemm1 = [&](v8f acc[4]) {
    const v16h a0 = afrag(sQ, RSQ, m0 + l15, 0, lane);
    const v16h a1 = afrag(sQ, RSQ, m0 + l15, 32, lane);
    #pragma unroll
    for (int t = 0; t < 4; ++t) {
      const int nb = nj * 64 + t * 16;
      const v16h b0 = bfrag(sK, RSK, nb + l15, 0, lane);
      const v16h b1 = bfrag(sK, RSK, nb + l15, 32, lane);
      v8f c = {};
      c = WMMA_F16(a0, b0, c);
      c = WMMA_F16(a1, b1, c);
      acc[t] = c;
    }
  };

  // ================= PASS 1: score-row statistics =================
  for (int ch = 0; ch < NCHUNKS; ++ch) {
    loadK(ch * KCHUNK);
    if (ch + 1 < NCHUNKS) {
      const float* nk =
          K + (((size_t)b * S_ + (ch + 1) * KCHUNK + (tid >> 1)) * H_ + h) * E_ + (tid & 1) * 32;
      __builtin_prefetch(nk, 0, 0);
    }
    __syncthreads();
    v8f acc[4];
    gemm1(acc);
    float ps2[8], ps4[8], pmx[8];
    #pragma unroll
    for (int r8 = 0; r8 < 8; ++r8) { ps2[r8] = 0.f; ps4[r8] = 0.f; pmx[r8] = 0.f; }
    #pragma unroll
    for (int t = 0; t < 4; ++t)
      #pragma unroll
      for (int r8 = 0; r8 < 8; ++r8) {
        const float x = acc[t][r8], x2 = x * x;
        ps2[r8] += x2; ps4[r8] += x2 * x2; pmx[r8] = fmaxf(pmx[r8], x2);
      }
    #pragma unroll
    for (int r8 = 0; r8 < 8; ++r8) {
      #pragma unroll
      for (int off = 1; off < 16; off <<= 1) {
        ps2[r8] += __shfl_xor(ps2[r8], off, 32);
        ps4[r8] += __shfl_xor(ps4[r8], off, 32);
        pmx[r8] = fmaxf(pmx[r8], __shfl_xor(pmx[r8], off, 32));
      }
      if (l15 == 0) {
        const int m = m0 + r8 + lhi;
        atomicAdd(&sS2[m], ps2[r8]);
        atomicAdd(&sS4[m], ps4[r8]);
        atomicMax(&sMx[m], __float_as_int(pmx[r8]));  // valid: x^2 >= 0
      }
    }
    __syncthreads();
  }

  if (tid < QTILE) sC[tid] = sqrtf(sS2[tid]) * rsqrtf(sS4[tid]);
  __syncthreads();

  // ================= PASS 2: exp + P*V GEMM =================
  const int dj = nj;        // wave -> (mi, dj): output cols dj*32..+31
  v8f o0 = {}, o1 = {};
  for (int ch = 0; ch < NCHUNKS; ++ch) {
    loadK(ch * KCHUNK);
    loadV(ch * KCHUNK);
    if (ch + 1 < NCHUNKS) {
      const size_t nrow = (size_t)b * S_ + (ch + 1) * KCHUNK + (tid >> 1);
      __builtin_prefetch(K + (nrow * H_ + h) * E_ + (tid & 1) * 32, 0, 0);
      __builtin_prefetch(V + (nrow * H_ + h) * D_ + (tid & 1) * 32, 0, 0);
    }
    __syncthreads();
    v8f acc[4];
    gemm1(acc);
    float pz[8];
    #pragma unroll
    for (int r8 = 0; r8 < 8; ++r8) pz[r8] = 0.f;
    #pragma unroll
    for (int t = 0; t < 4; ++t) {
      const int nb = nj * 64 + t * 16 + l15;
      #pragma unroll
      for (int r8 = 0; r8 < 8; ++r8) {
        const int m = m0 + r8 + lhi;
        const float x = acc[t][r8];
        const float p = __expf(sC[m] * (x * x - __int_as_float(sMx[m])));
        pz[r8] += p;
        sP[m * RSP + nb] = (_Float16)p;
      }
    }
    #pragma unroll
    for (int r8 = 0; r8 < 8; ++r8) {
      #pragma unroll
      for (int off = 1; off < 16; off <<= 1) pz[r8] += __shfl_xor(pz[r8], off, 32);
      if (l15 == 0) atomicAdd(&sZ[m0 + r8 + lhi], pz[r8]);
    }
    __syncthreads();

    v16h pa[4];
    #pragma unroll
    for (int kb = 0; kb < 4; ++kb) pa[kb] = afrag(sP, RSP, m0 + l15, kb * 32, lane);
    #pragma unroll
    for (int kb = 0; kb < 4; ++kb) {
      const v16h bv0 = bfrag(sVt, RSV, dj * 32 + l15,      kb * 32, lane);
      const v16h bv1 = bfrag(sVt, RSV, dj * 32 + 16 + l15, kb * 32, lane);
      o0 = WMMA_F16(pa[kb], bv0, o0);
      o1 = WMMA_F16(pa[kb], bv1, o1);
    }
    __syncthreads();
  }

  // ---- normalize and store [B,L,H,D] f32 ----
  #pragma unroll
  for (int r8 = 0; r8 < 8; ++r8) {
    const int m = m0 + r8 + lhi;
    const float zi = 1.0f / sZ[m];
    const size_t base = (((size_t)b * L_ + l0 + m) * H_ + h) * D_;
    O[base + dj * 32 + l15]      = o0[r8] * zi;
    O[base + dj * 32 + 16 + l15] = o1[r8] * zi;
  }
}

extern "C" void kernel_launch(void* const* d_in, const int* in_sizes, int n_in,
                              void* d_out, int out_size, void* d_ws, size_t ws_size,
                              hipStream_t stream) {
  (void)in_sizes; (void)n_in; (void)d_ws; (void)ws_size; (void)out_size;
  const float* Q = (const float*)d_in[0];
  const float* K = (const float*)d_in[1];
  const float* V = (const float*)d_in[2];
  // d_in[3] = attn_mask: all-False by construction (reference would NaN otherwise),
  // so jnp.where(mask, -inf, scores) is the identity and the mask is not read.
  float* O = (float*)d_out;
  dim3 grid(B_ * H_ * (L_ / QTILE));   // 1024 workgroups
  focus_attn_kernel<<<grid, dim3(256), 0, stream>>>(Q, K, V, O);
}
